// MultiDimWindowTransformerLayer_1494648619773
// MI455X (gfx1250) — matrix operand order repaired and checked by reading, not compiled, not run on hardware
//
#include <hip/hip_runtime.h>
#include <hip/hip_bf16.h>
#include <math.h>

typedef __attribute__((ext_vector_type(16))) _Float16 v16h;
typedef __attribute__((ext_vector_type(8)))  _Float16 v8h;
typedef __attribute__((ext_vector_type(8)))  float    v8f;

#define L_   2048
#define C_   8
#define H_   256
#define NH_  8
#define HD_  32
#define D_   256
#define FF_  1024
#define WIN_ 21
#define WC_  (WIN_ * C_)

// ---------------------------------------------------------------------------
// CDNA5 async copy: 16B per lane, global -> LDS, tracked by ASYNCcnt.
// lds = wave-relative LDS byte offset (low 32 bits of generic shared ptr).
// ---------------------------------------------------------------------------
__device__ __forceinline__ void async_lds_b128(unsigned lds, const _Float16* g)
{
    asm volatile("global_load_async_to_lds_b128 %0, %1, off"
                 :: "v"(lds), "v"(g)
                 : "memory");
}
__device__ __forceinline__ void wait_async0()
{
    asm volatile("s_wait_asynccnt 0" ::: "memory");
}

// ---------------------------------------------------------------------------
// WMMA GEMM, all-f16 operands: C[M,N] = A[M,K] @ B[K,N] + bias.
//   A : M x K row-major f16 (fragment = two contiguous 16B runs -> b128 loads)
//   BT: N x K row-major f16 (B pre-transposed; fragment = two 16B runs, like A)
// Block: 256 threads = 8 waves. Wave w computes rows [m0, m0+16) x 64 cols
// with 4 accumulators (A fragment reused 4x). BT tile (64x32 f16) staged in
// LDS via async global->LDS copies, double-buffered so the next tile's fill
// overlaps the current tile's 4 WMMAs.
// Requires M % 128 == 0, N % 64 == 0, K % 32 == 0 (true at all call sites).
// ---------------------------------------------------------------------------
__global__ __launch_bounds__(256) void gemm_wmma_h(
    const _Float16* __restrict__ A,
    const _Float16* __restrict__ BT,
    const float*    __restrict__ bias,
    float*          __restrict__ Cf,   // f32 output (used when Ch == nullptr)
    _Float16*       __restrict__ Ch,   // f16 output (optional)
    int M, int N, int K, int relu)
{
    __shared__ _Float16 bs[2][64][40];  // double-buffered 64 cols x 32 K, padded

    const int lane = threadIdx.x & 31;
    const int wave = threadIdx.x >> 5;
    const int m0   = blockIdx.y * 128 + wave * 16;
    const int n0b  = blockIdx.x * 64;

    const int arow  = m0 + (lane & 15);
    const int akoff = (lane >> 4) * 8;      // ISA A layout: lanes>=16 start K+8
    const int kloc  = (lane >> 4) * 8;      // ISA B layout: lanes>=16 start K+8

    const int scol = threadIdx.x >> 2;      // 0..63 staging column
    const int sch  = (threadIdx.x & 3) * 8; // 8-element (16B) chunk

    const _Float16* gsrc = BT + (size_t)(n0b + scol) * K + sch;
    const unsigned  lds0 = (unsigned)(size_t)&bs[0][scol][sch];
    const unsigned  lds1 = (unsigned)(size_t)&bs[1][scol][sch];

    v8f acc[4] = {};

    // prologue: fill buffer 0 with the first K tile
    async_lds_b128(lds0, gsrc);
    wait_async0();
    __syncthreads();

    int cur = 0;
    for (int k0 = 0; k0 < K; k0 += 32) {
        // kick off the next tile's async fill into the other buffer
        if (k0 + 32 < K) {
            async_lds_b128(cur ? lds0 : lds1, gsrc + k0 + 32);
            __builtin_prefetch(A + (size_t)arow * K + (k0 + 32) + akoff, 0, 1);
        }

        // A fragment: two contiguous 16B runs
        v16h a;
        const _Float16* ap = A + (size_t)arow * K + k0 + akoff;
        #pragma unroll
        for (int j = 0; j < 16; ++j)        // K(j) = (j<8 ? j : j+8) + akoff
            a[j] = ap[j < 8 ? j : j + 8];

        // all four B fragments up front, then one WMMA burst
        v16h b0, b1, b2, b3;
        {
            const _Float16* p0 = &bs[cur][ 0 + (lane & 15)][kloc];
            const _Float16* p1 = &bs[cur][16 + (lane & 15)][kloc];
            const _Float16* p2 = &bs[cur][32 + (lane & 15)][kloc];
            const _Float16* p3 = &bs[cur][48 + (lane & 15)][kloc];
            #pragma unroll
            for (int j = 0; j < 16; ++j) {
                const int jj = j < 8 ? j : j + 8;
                b0[j] = p0[jj]; b1[j] = p1[jj]; b2[j] = p2[jj]; b3[j] = p3[jj];
            }
        }
        acc[0] = __builtin_amdgcn_wmma_f32_16x16x32_f16(false, a, false, b0, (short)0, acc[0], false, false);
        acc[1] = __builtin_amdgcn_wmma_f32_16x16x32_f16(false, a, false, b1, (short)0, acc[1], false, false);
        acc[2] = __builtin_amdgcn_wmma_f32_16x16x32_f16(false, a, false, b2, (short)0, acc[2], false, false);
        acc[3] = __builtin_amdgcn_wmma_f32_16x16x32_f16(false, a, false, b3, (short)0, acc[3], false, false);

        // drain this wave's async fill, then rendezvous before buffer swap
        wait_async0();
        __syncthreads();
        cur ^= 1;
    }

    const int rbase = m0 + ((lane >> 4) * 8);   // ISA C layout
    #pragma unroll
    for (int nt = 0; nt < 4; ++nt) {
        const int col  = n0b + nt * 16 + (lane & 15);
        const float bv = bias[col];
        #pragma unroll
        for (int i = 0; i < 8; ++i) {
            float v = acc[nt][i] + bv;
            if (relu) v = fmaxf(v, 0.0f);
            if (Ch) Ch[(size_t)(rbase + i) * N + col] = (_Float16)v;
            else    Cf[(size_t)(rbase + i) * N + col] = v;
        }
    }
}

// ---------------------------------------------------------------------------
// One-time f32 -> f16 conversion (flat).
// ---------------------------------------------------------------------------
__global__ __launch_bounds__(256) void f32_to_f16_kernel(const float* __restrict__ in,
                                                         _Float16* __restrict__ out)
{
    const size_t i = (size_t)blockIdx.x * 256 + threadIdx.x;
    out[i] = (_Float16)in[i];
}

// ---------------------------------------------------------------------------
// Weight transpose+convert: in[K,N] f32 -> out[N,K] f16 (coalesced writes).
// ---------------------------------------------------------------------------
__global__ __launch_bounds__(256) void transpose_f16_kernel(const float* __restrict__ in,
                                                            _Float16* __restrict__ out,
                                                            int K, int N)
{
    const size_t o = (size_t)blockIdx.x * 256 + threadIdx.x;  // o = n*K + k
    const int nn = (int)(o / K);
    const int kk = (int)(o - (size_t)nn * K);
    out[o] = (_Float16)in[(size_t)kk * N + nn];
}

// ---------------------------------------------------------------------------
// Relative-position sinusoid embedding, written directly as f16 GEMM input.
// ---------------------------------------------------------------------------
__global__ __launch_bounds__(256) void relpos_kernel(const int* __restrict__ pos,
                                                     _Float16* __restrict__ r)
{
    const int idx = blockIdx.x;            // w*L + l
    const int h   = threadIdx.x;
    const int w   = idx / L_;
    const int l   = idx - w * L_;
    const int tt  = w + l;
    const float pl     = (float)pos[l];
    const float padded = (tt >= 10 && tt < L_ + 10) ? (float)pos[tt - 10] : 0.0f;
    const float relp   = padded - pl;
    const float invf   = __powf(10000.0f, -(float)((h >> 1) * 2) / 256.0f);
    const float ang    = relp * invf;
    r[(size_t)idx * H_ + h] = (_Float16)((h & 1) ? __cosf(ang) : __sinf(ang));
}

// ---------------------------------------------------------------------------
// Windowed multi-dim attention. One block per position l; wave n = head n;
// lane t = feature h (HD==32==wave32). Scores in LDS, softmax via wave32
// shuffle butterflies. Output written as f16 (feeds out-proj WMMA GEMM).
// ---------------------------------------------------------------------------
__global__ __launch_bounds__(256) void attn_kernel(const float* __restrict__ qkv,
                                                   const float* __restrict__ re,
                                                   const float* __restrict__ brw,
                                                   const float* __restrict__ brr,
                                                   _Float16* __restrict__ z)
{
    __shared__ float qw[NH_][C_][HD_];
    __shared__ float qr[NH_][C_][HD_];
    __shared__ float sc[NH_][C_][WC_];
    __shared__ float bd[NH_][C_][WIN_];

    const int l = blockIdx.x;
    const int n = threadIdx.x >> 5;
    const int t = threadIdx.x & 31;
    const float scale = 0.17677669529663687f;   // HD^-0.5

    const float bw = brw[n * HD_ + t];
    const float br = brr[n * HD_ + t];
    #pragma unroll
    for (int c = 0; c < C_; ++c) {
        float qv = qkv[(size_t)(l * C_ + c) * 768 + n * HD_ + t] * scale;
        qw[n][c][t] = qv + bw;
        qr[n][c][t] = qv + br;
    }
    __syncthreads();

    // BD[c][w] = (q+brr) . re[w, l, n, :]
    for (int idx = t; idx < C_ * WIN_; idx += 32) {
        int c = idx / WIN_, w = idx % WIN_;
        const float* rp = re + (size_t)(w * L_ + l) * D_ + n * HD_;
        float s = 0.0f;
        #pragma unroll
        for (int h = 0; h < HD_; ++h) s += qr[n][c][h] * rp[h];
        bd[n][c][w] = s;
    }
    __syncthreads();

    // scores[c][w*C+d] = (q+brw) . k[l+w-10, d, n, :] + BD[c][w]  (masked)
    for (int idx = t; idx < C_ * WC_; idx += 32) {
        int c = idx / WC_, rem = idx % WC_;
        int w = rem / C_, d = rem % C_;
        int lw = l + w - 10;
        float s;
        if (lw >= 0 && lw < L_) {
            const float* kp = qkv + (size_t)(lw * C_ + d) * 768 + 256 + n * HD_;
            s = bd[n][c][w];
            #pragma unroll
            for (int h = 0; h < HD_; ++h) s += qw[n][c][h] * kp[h];
        } else {
            s = -3.0e38f;
        }
        sc[n][c][rem] = s;
    }
    __syncthreads();

    // softmax over the 168 (w,d) entries per c
    float invs[C_];
    #pragma unroll
    for (int c = 0; c < C_; ++c) {
        float m = -3.0e38f;
        for (int j = t; j < WC_; j += 32) m = fmaxf(m, sc[n][c][j]);
        for (int s1 = 16; s1 > 0; s1 >>= 1) m = fmaxf(m, __shfl_xor(m, s1, 32));
        float sum = 0.0f;
        for (int j = t; j < WC_; j += 32) {
            float e = __expf(sc[n][c][j] - m);
            sc[n][c][j] = e;
            sum += e;
        }
        for (int s1 = 16; s1 > 0; s1 >>= 1) sum += __shfl_xor(sum, s1, 32);
        invs[c] = 1.0f / sum;
    }
    __syncthreads();

    // z[c][h] = sum_j A[c][j] * v[lw(j), d(j), n, h]
    float acc[C_] = {};
    for (int j = 0; j < WC_; ++j) {
        int w = j / C_, d = j % C_;
        int lw = l + w - 10;
        if (lw < 0 || lw >= L_) continue;
        float vv = qkv[(size_t)(lw * C_ + d) * 768 + 512 + n * HD_ + t];
        #pragma unroll
        for (int c = 0; c < C_; ++c) acc[c] += sc[n][c][j] * vv;
    }
    #pragma unroll
    for (int c = 0; c < C_; ++c)
        z[(size_t)(l * C_ + c) * D_ + n * HD_ + t] = (_Float16)(acc[c] * invs[c]);
}

// ---------------------------------------------------------------------------
// out = LayerNorm(x + y) * g + b, one 256-wide row per block.
// Optionally also writes an f16 copy (GEMM input for the next stage).
// ---------------------------------------------------------------------------
__global__ __launch_bounds__(256) void add_ln_kernel(const float* __restrict__ x,
                                                     const float* __restrict__ y,
                                                     const float* __restrict__ g,
                                                     const float* __restrict__ b,
                                                     float* __restrict__ out,
                                                     _Float16* __restrict__ outh)
{
    __shared__ float red[256];
    const int row = blockIdx.x;
    const int t   = threadIdx.x;
    const float v = x[(size_t)row * H_ + t] + y[(size_t)row * H_ + t];

    red[t] = v;
    __syncthreads();
    for (int s = 128; s > 0; s >>= 1) { if (t < s) red[t] += red[t + s]; __syncthreads(); }
    const float mu = red[0] * (1.0f / 256.0f);
    __syncthreads();

    const float dv = v - mu;
    red[t] = dv * dv;
    __syncthreads();
    for (int s = 128; s > 0; s >>= 1) { if (t < s) red[t] += red[t + s]; __syncthreads(); }
    const float var = red[0] * (1.0f / 256.0f);
    const float rs  = rsqrtf(var + 1e-5f);

    const float o = dv * rs * g[t] + b[t];
    out[(size_t)row * H_ + t] = o;
    if (outh) outh[(size_t)row * H_ + t] = (_Float16)o;
}

// ---------------------------------------------------------------------------
extern "C" void kernel_launch(void* const* d_in, const int* in_sizes, int n_in,
                              void* d_out, int out_size, void* d_ws, size_t ws_size,
                              hipStream_t stream)
{
    const float* x    = (const float*)d_in[0];
    const int*   pos  = (const int*)  d_in[1];
    const float* Wqkv = (const float*)d_in[2];
    const float* bqkv = (const float*)d_in[3];
    const float* We   = (const float*)d_in[4];
    const float* be   = (const float*)d_in[5];
    const float* brw  = (const float*)d_in[6];
    const float* brr  = (const float*)d_in[7];
    const float* Wout = (const float*)d_in[8];
    const float* bout = (const float*)d_in[9];
    const float* ln1g = (const float*)d_in[10];
    const float* ln1b = (const float*)d_in[11];
    const float* ln2g = (const float*)d_in[12];
    const float* ln2b = (const float*)d_in[13];
    const float* Wff1 = (const float*)d_in[14];
    const float* bff1 = (const float*)d_in[15];
    const float* Wff2 = (const float*)d_in[16];
    const float* bff2 = (const float*)d_in[17];

    const int M  = L_ * C_;      // 16384 activation rows
    const int MR = WIN_ * L_;    // 43008 relpos rows

    // Workspace carve-up (256B-aligned regions).
    char*  w   = (char*)d_ws;
    size_t off = 0;
    auto alloc = [&](size_t bytes) -> void* {
        void* p = w + off;
        off = (off + bytes + 255) & ~(size_t)255;
        return p;
    };
    float*    qkvb  = (float*)   alloc((size_t)M * 768 * 4);   // f32 qkv
    float*    reb   = (float*)   alloc((size_t)MR * D_ * 4);   // f32 re (attn_out aliases)
    float*    x1b   = (float*)   alloc((size_t)M * H_ * 4);    // f32 LN1 out
    _Float16* xh    = (_Float16*)alloc((size_t)M * H_ * 2);
    _Float16* rh    = (_Float16*)alloc((size_t)MR * H_ * 2);
    _Float16* zh    = (_Float16*)alloc((size_t)M * D_ * 2);
    _Float16* x1h   = (_Float16*)alloc((size_t)M * H_ * 2);
    _Float16* ff1h  = (_Float16*)alloc((size_t)M * FF_ * 2);
    _Float16* WqkvT = (_Float16*)alloc((size_t)768 * H_ * 2);
    _Float16* WeT   = (_Float16*)alloc((size_t)D_ * H_ * 2);
    _Float16* WoutT = (_Float16*)alloc((size_t)H_ * D_ * 2);
    _Float16* Wff1T = (_Float16*)alloc((size_t)FF_ * H_ * 2);
    _Float16* Wff2T = (_Float16*)alloc((size_t)H_ * FF_ * 2);
    float*    attnb = reb;    // re dead after attention
    float*    ffob  = qkvb;   // qkv dead after attention

    // 0) one-time f16 conversions / weight transposes
    f32_to_f16_kernel<<<dim3(M * H_ / 256), 256, 0, stream>>>(x, xh);
    transpose_f16_kernel<<<dim3(768 * H_ / 256), 256, 0, stream>>>(Wqkv, WqkvT, H_, 768);
    transpose_f16_kernel<<<dim3(D_ * H_ / 256), 256, 0, stream>>>(We,   WeT,   H_, D_);
    transpose_f16_kernel<<<dim3(H_ * D_ / 256), 256, 0, stream>>>(Wout, WoutT, D_, H_);
    transpose_f16_kernel<<<dim3(FF_ * H_ / 256), 256, 0, stream>>>(Wff1, Wff1T, H_, FF_);
    transpose_f16_kernel<<<dim3(H_ * FF_ / 256), 256, 0, stream>>>(Wff2, Wff2T, FF_, H_);

    // 1) qkv = x @ Wqkv + bqkv                    (f32 out, feeds attention)
    gemm_wmma_h<<<dim3(768 / 64, M / 128), 256, 0, stream>>>(
        xh, WqkvT, bqkv, qkvb, nullptr, M, 768, H_, 0);
    // 2) r = relpos sinusoids (f16)
    relpos_kernel<<<dim3(MR), 256, 0, stream>>>(pos, rh);
    // 3) re = r @ We + be                         (f32 out, feeds attention)
    gemm_wmma_h<<<dim3(D_ / 64, MR / 128), 256, 0, stream>>>(
        rh, WeT, be, reb, nullptr, MR, D_, H_, 0);
    // 4) windowed attention -> z (f16)
    attn_kernel<<<dim3(L_), 256, 0, stream>>>(qkvb, reb, brw, brr, zh);
    // 5) attn_out = z @ Wout + bout               (f32 out, feeds LN1)
    gemm_wmma_h<<<dim3(H_ / 64, M / 128), 256, 0, stream>>>(
        zh, WoutT, bout, attnb, nullptr, M, H_, D_, 0);
    // 6) x1 = LN1(x + attn_out)                   (f32 + f16 copies)
    add_ln_kernel<<<dim3(M), 256, 0, stream>>>(x, attnb, ln1g, ln1b, x1b, x1h);
    // 7) ff1 = relu(x1 @ Wff1 + bff1)             (f16 out, feeds ff2 GEMM)
    gemm_wmma_h<<<dim3(FF_ / 64, M / 128), 256, 0, stream>>>(
        x1h, Wff1T, bff1, nullptr, ff1h, M, FF_, H_, 1);
    // 8) ffout = ff1 @ Wff2 + bff2                (f32 out, feeds LN2)
    gemm_wmma_h<<<dim3(H_ / 64, M / 128), 256, 0, stream>>>(
        ff1h, Wff2T, bff2, ffob, nullptr, M, H_, FF_, 0);
    // 9) out = LN2(x1 + ffout)
    add_ln_kernel<<<dim3(M), 256, 0, stream>>>(x1b, ffob, ln2g, ln2b, (float*)d_out, nullptr);
}
